// TensorParallelTransformerBlock_68212670595280
// MI455X (gfx1250) — matrix operand order repaired and checked by reading, not compiled, not run on hardware
//
#include <hip/hip_runtime.h>

#define DEV __device__ __forceinline__

typedef __attribute__((ext_vector_type(16))) __bf16       v16bf;
typedef __attribute__((ext_vector_type(8)))  float        v8f;
typedef __attribute__((ext_vector_type(4)))  unsigned int u32x4;

union Frag { v16bf v; u32x4 q[2]; };

// Problem dims (hardcoded from reference)
constexpr int Dm    = 2048;   // hidden
constexpr int Hn    = 16;     // heads
constexpr int HDm   = 128;    // head dim
constexpr int FFm   = 8192;   // ffn dim
constexpr int Bm    = 4;      // batch
constexpr int Sm    = 2048;   // seq
constexpr int Mrows = Bm * Sm; // 8192 token rows

DEV unsigned short f2b(float f) {
    union { float f; unsigned u; } c; c.f = f;
    unsigned u = c.u;
    return (unsigned short)((u + 0x7FFFu + ((u >> 16) & 1u)) >> 16);
}

DEV v8f wmma_bf16(const Frag& a, const Frag& b, v8f c) {
    // D = A(16x32) * B(32x16) + C, f32 accum
    return __builtin_amdgcn_wmma_f32_16x16x32_bf16(
        false, a.v, false, b.v, (short)0, c, false, false);
}

// ---------------------------------------------------------------------------
// Weight convert + transpose: out[n*K + k] = bf16(in[k*N + n])
// ---------------------------------------------------------------------------
__global__ __launch_bounds__(256)
void wt_bf16(const float* __restrict__ in, unsigned short* __restrict__ out,
             int K, int N) {
    size_t idx = (size_t)blockIdx.x * 256 + threadIdx.x;
    if (idx >= (size_t)K * N) return;
    int n = (int)(idx / K);
    int k = (int)(idx % K);
    out[idx] = f2b(in[(size_t)k * N + n]);
}

// ---------------------------------------------------------------------------
// LayerNorm over D=2048, one block per row, 8 elems/thread, bf16 out
// ---------------------------------------------------------------------------
__global__ __launch_bounds__(256)
void ln_kernel(const float* __restrict__ x, const float* __restrict__ g,
               const float* __restrict__ bta, unsigned short* __restrict__ out) {
    __shared__ float red[8];
    const int rid = blockIdx.x, tid = threadIdx.x;
    const int lane = tid & 31, w = tid >> 5;
    const float* row = x + (size_t)rid * Dm;
    float v[8];
    float s = 0.f;
#pragma unroll
    for (int j = 0; j < 8; ++j) { v[j] = row[tid + j * 256]; s += v[j]; }
#pragma unroll
    for (int off = 16; off; off >>= 1) s += __shfl_xor(s, off, 32);
    if (lane == 0) red[w] = s;
    __syncthreads();
    float tot = 0.f;
#pragma unroll
    for (int j = 0; j < 8; ++j) tot += red[j];
    const float mu = tot * (1.0f / Dm);
    __syncthreads();
    float s2 = 0.f;
#pragma unroll
    for (int j = 0; j < 8; ++j) { float d = v[j] - mu; s2 += d * d; }
#pragma unroll
    for (int off = 16; off; off >>= 1) s2 += __shfl_xor(s2, off, 32);
    if (lane == 0) red[w] = s2;
    __syncthreads();
    tot = 0.f;
#pragma unroll
    for (int j = 0; j < 8; ++j) tot += red[j];
    const float rs = rsqrtf(tot * (1.0f / Dm) + 1e-5f);
#pragma unroll
    for (int j = 0; j < 8; ++j) {
        int c = tid + j * 256;
        out[(size_t)rid * Dm + c] = f2b((v[j] - mu) * rs * g[c] + bta[c]);
    }
}

// ---------------------------------------------------------------------------
// bf16 GEMM: C[M,N] = A[M,K] * Bt[N,K]^T  (Bt row-major = B^T)
// 128x128 block tile, BK=32, double-buffered LDS (1 barrier/iter) + prefetch.
// 8 waves, wave tile 64x32 (4x2 WMMA tiles).
// MODE 1: head-split bf16 store (Q,K)    [b,h,s,hd]
// MODE 2: head-split transposed (V)      [b,h,hd,s]
// MODE 3: fp32 + residual                outf = resid + acc
// MODE 4: ReLU bf16 store                row-major
// ---------------------------------------------------------------------------
template <int MODE>
__global__ __launch_bounds__(256)
void gemm_bf16(const unsigned short* __restrict__ A,
               const unsigned short* __restrict__ Bt,
               int Ndim, int Kdim,
               const float* __restrict__ resid,
               unsigned short* __restrict__ outb,
               float* __restrict__ outf) {
    __shared__ unsigned short sA[2][128 * 32];
    __shared__ unsigned short sB[2][128 * 32];
    const int tid  = threadIdx.x;
    const int lane = tid & 31, w = tid >> 5;
    const int half = lane >> 4, ln = lane & 15;
    const int wm = w >> 2, wn = w & 3;          // wave grid 2x4
    const int m0 = blockIdx.y * 128, n0 = blockIdx.x * 128;

    const v8f vz = {0.f, 0.f, 0.f, 0.f, 0.f, 0.f, 0.f, 0.f};
    v8f acc[4][2];
#pragma unroll
    for (int i = 0; i < 4; ++i)
#pragma unroll
        for (int j = 0; j < 2; ++j) acc[i][j] = vz;

    const int lr = tid >> 2;          // 0..63
    const int lc = (tid & 3) * 8;     // 0,8,16,24
    const unsigned short* Ar0 = A  + (size_t)(m0 + lr)      * Kdim + lc;
    const unsigned short* Ar1 = A  + (size_t)(m0 + lr + 64) * Kdim + lc;
    const unsigned short* Br0 = Bt + (size_t)(n0 + lr)      * Kdim + lc;
    const unsigned short* Br1 = Bt + (size_t)(n0 + lr + 64) * Kdim + lc;
    const int sAo0 = lr * 32 + lc, sAo1 = (lr + 64) * 32 + lc;

    // preload tile 0
    *(u32x4*)&sA[0][sAo0] = *(const u32x4*)Ar0;
    *(u32x4*)&sA[0][sAo1] = *(const u32x4*)Ar1;
    *(u32x4*)&sB[0][sAo0] = *(const u32x4*)Br0;
    *(u32x4*)&sB[0][sAo1] = *(const u32x4*)Br1;
    __syncthreads();

    int cur = 0;
    for (int kk = 0; kk < Kdim; kk += 32) {
        const int nxt = cur ^ 1;
        if (kk + 32 < Kdim) {   // stage next tile while computing this one
            *(u32x4*)&sA[nxt][sAo0] = *(const u32x4*)(Ar0 + kk + 32);
            *(u32x4*)&sA[nxt][sAo1] = *(const u32x4*)(Ar1 + kk + 32);
            *(u32x4*)&sB[nxt][sAo0] = *(const u32x4*)(Br0 + kk + 32);
            *(u32x4*)&sB[nxt][sAo1] = *(const u32x4*)(Br1 + kk + 32);
            if (kk + 64 < Kdim) {  // prefetch tile after next into caches
                __builtin_prefetch(Ar0 + kk + 64, 0, 3);
                __builtin_prefetch(Ar1 + kk + 64, 0, 3);
                __builtin_prefetch(Br0 + kk + 64, 0, 3);
                __builtin_prefetch(Br1 + kk + 64, 0, 3);
            }
        }

        Frag af[4], bf[2];
#pragma unroll
        for (int mi = 0; mi < 4; ++mi) {
            int r = wm * 64 + mi * 16 + ln;
            af[mi].q[0] = *(const u32x4*)&sA[cur][r * 32 + half * 8];
            af[mi].q[1] = *(const u32x4*)&sA[cur][r * 32 + 16 + half * 8];
        }
#pragma unroll
        for (int ni = 0; ni < 2; ++ni) {
            int c = wn * 32 + ni * 16 + ln;
            bf[ni].q[0] = *(const u32x4*)&sB[cur][c * 32 + half * 8];
            bf[ni].q[1] = *(const u32x4*)&sB[cur][c * 32 + 16 + half * 8];
        }
#pragma unroll
        for (int mi = 0; mi < 4; ++mi)
#pragma unroll
            for (int ni = 0; ni < 2; ++ni)
                acc[mi][ni] = wmma_bf16(af[mi], bf[ni], acc[mi][ni]);

        __syncthreads();
        cur = nxt;
    }

#pragma unroll
    for (int mi = 0; mi < 4; ++mi)
#pragma unroll
        for (int ni = 0; ni < 2; ++ni) {
            const int rowb = m0 + wm * 64 + mi * 16 + half * 8;
            const int col  = n0 + wn * 32 + ni * 16 + ln;
#pragma unroll
            for (int i = 0; i < 8; ++i) {
                const float vv = acc[mi][ni][i];
                const int r = rowb + i;
                if constexpr (MODE == 1) {
                    int bI = r >> 11, sI = r & (Sm - 1);
                    int hI = col >> 7, hdI = col & (HDm - 1);
                    outb[(((size_t)(bI * Hn + hI)) * Sm + sI) * HDm + hdI] = f2b(vv);
                } else if constexpr (MODE == 2) {
                    int bI = r >> 11, sI = r & (Sm - 1);
                    int hI = col >> 7, hdI = col & (HDm - 1);
                    outb[(((size_t)(bI * Hn + hI)) * HDm + hdI) * Sm + sI] = f2b(vv);
                } else if constexpr (MODE == 3) {
                    size_t idx = (size_t)r * Ndim + col;
                    outf[idx] = resid[idx] + vv;
                } else {
                    outb[(size_t)r * Ndim + col] = f2b(vv > 0.f ? vv : 0.f);
                }
            }
        }
}

// ---------------------------------------------------------------------------
// Flash attention: grid (S/128, B*H); 8 waves x 16 queries; 32-key chunks.
// K/V chunks staged cooperatively into double-buffered LDS (shared by all
// 8 waves -> 8x less global traffic), 1 barrier per chunk.
// q,k: [b,h,s,hd] bf16;  vt: [b,h,hd,s] bf16;  out: [b,s,d] bf16
// ---------------------------------------------------------------------------
__global__ __launch_bounds__(256)
void attn_kernel(const unsigned short* __restrict__ qg,
                 const unsigned short* __restrict__ kg,
                 const unsigned short* __restrict__ vt,
                 unsigned short* __restrict__ ao) {
    __shared__ unsigned short sK[2][32 * 128];   // [key][hd]
    __shared__ unsigned short sV[2][128 * 32];   // [hd][key]
    __shared__ unsigned short sP[8 * 16 * 32];   // per-wave 16x32 P scratch
    const int tid  = threadIdx.x;
    const int lane = tid & 31, w = tid >> 5;
    const int half = lane >> 4, ln = lane & 15;
    const int bh = blockIdx.y;                  // b*H + h
    const int bI = bh >> 4, hI = bh & (Hn - 1);
    const int q0 = blockIdx.x * 128 + w * 16;

    // cooperative K/V chunk staging (32 keys)
    const int kr = tid >> 4, kc2 = (tid & 15) * 8;   // K: 2 rows of 16x(8 bf16)
    const int vr = tid >> 2, vc = (tid & 3) * 8;     // V: 2 rows of 64x(8 bf16)
    auto stage = [&](int buf, int key0) {
        *(u32x4*)&sK[buf][kr * 128 + kc2] =
            *(const u32x4*)&kg[((size_t)bh * Sm + key0 + kr) * HDm + kc2];
        *(u32x4*)&sK[buf][(kr + 16) * 128 + kc2] =
            *(const u32x4*)&kg[((size_t)bh * Sm + key0 + kr + 16) * HDm + kc2];
        *(u32x4*)&sV[buf][vr * 32 + vc] =
            *(const u32x4*)&vt[((size_t)bh * HDm + vr) * Sm + key0 + vc];
        *(u32x4*)&sV[buf][(vr + 64) * 32 + vc] =
            *(const u32x4*)&vt[((size_t)bh * HDm + vr + 64) * Sm + key0 + vc];
    };

    // Load Q fragments for all 4 K-chunks of HD=128 (A-matrix layout)
    Frag qf[4];
#pragma unroll
    for (int kc = 0; kc < 4; ++kc) {
        const unsigned short* qp =
            qg + ((size_t)bh * Sm + q0 + ln) * HDm + kc * 32 + half * 8;
        qf[kc].q[0] = *(const u32x4*)qp;
        qf[kc].q[1] = *(const u32x4*)(qp + 16);
    }

    const v8f vz = {0.f, 0.f, 0.f, 0.f, 0.f, 0.f, 0.f, 0.f};
    v8f o[8];
#pragma unroll
    for (int nt = 0; nt < 8; ++nt) o[nt] = vz;
    float m_i[8], l_i[8];
#pragma unroll
    for (int i = 0; i < 8; ++i) { m_i[i] = -3.0e38f; l_i[i] = 0.f; }

    unsigned short* sp = &sP[w * 512];
    const float sc = 0.08838834764831845f;      // 1/sqrt(128)

    stage(0, 0);
    __syncthreads();

    int cur = 0;
    for (int kt = 0; kt < Sm / 32; ++kt) {
        if (kt + 1 < Sm / 32) stage(cur ^ 1, (kt + 1) * 32);

        // --- scores S = Q K^T  (16 x 32) from LDS ---
        v8f s0 = vz, s1 = vz;
#pragma unroll
        for (int kc = 0; kc < 4; ++kc) {
            Frag b0, b1;
            b0.q[0] = *(const u32x4*)&sK[cur][ln * 128 + kc * 32 + half * 8];
            b0.q[1] = *(const u32x4*)&sK[cur][ln * 128 + kc * 32 + 16 + half * 8];
            b1.q[0] = *(const u32x4*)&sK[cur][(16 + ln) * 128 + kc * 32 + half * 8];
            b1.q[1] = *(const u32x4*)&sK[cur][(16 + ln) * 128 + kc * 32 + 16 + half * 8];
            s0 = wmma_bf16(qf[kc], b0, s0);
            s1 = wmma_bf16(qf[kc], b1, s1);
        }
        // --- streaming softmax (row = half*8 + i, spread over 16 lanes) ---
        float p0[8], p1[8], alpha[8];
#pragma unroll
        for (int i = 0; i < 8; ++i) {
            float a = s0[i] * sc, b = s1[i] * sc;
            s0[i] = a; s1[i] = b;
            float mx = fmaxf(a, b);
#pragma unroll
            for (int off = 1; off < 16; off <<= 1)
                mx = fmaxf(mx, __shfl_xor(mx, off, 32));
            float mn = fmaxf(m_i[i], mx);
            alpha[i] = __expf(m_i[i] - mn);
            p0[i] = __expf(s0[i] - mn);
            p1[i] = __expf(s1[i] - mn);
            float rsum = p0[i] + p1[i];
#pragma unroll
            for (int off = 1; off < 16; off <<= 1)
                rsum += __shfl_xor(rsum, off, 32);
            l_i[i] = l_i[i] * alpha[i] + rsum;
            m_i[i] = mn;
        }
#pragma unroll
        for (int nt = 0; nt < 8; ++nt)
#pragma unroll
            for (int i = 0; i < 8; ++i) o[nt][i] *= alpha[i];

        // --- P: C-layout -> A-layout via wave-private LDS ---
#pragma unroll
        for (int i = 0; i < 8; ++i) {
            sp[(half * 8 + i) * 32 + ln]      = f2b(p0[i]);
            sp[(half * 8 + i) * 32 + 16 + ln] = f2b(p1[i]);
        }
        asm volatile("s_wait_dscnt 0" ::: "memory");  // same-wave LDS RAW
        Frag pf;
        pf.q[0] = *(const u32x4*)&sp[ln * 32 + half * 8];
        pf.q[1] = *(const u32x4*)&sp[ln * 32 + 16 + half * 8];

        // --- O += P * V from LDS (V^T rows are key-contiguous) ---
#pragma unroll
        for (int nt = 0; nt < 8; ++nt) {
            Frag vf;
            vf.q[0] = *(const u32x4*)&sV[cur][(nt * 16 + ln) * 32 + half * 8];
            vf.q[1] = *(const u32x4*)&sV[cur][(nt * 16 + ln) * 32 + 16 + half * 8];
            o[nt] = wmma_bf16(pf, vf, o[nt]);
        }

        __syncthreads();
        cur ^= 1;
    }

    float inv[8];
#pragma unroll
    for (int i = 0; i < 8; ++i) inv[i] = 1.0f / l_i[i];
#pragma unroll
    for (int nt = 0; nt < 8; ++nt) {
        const int dcol = hI * HDm + nt * 16 + ln;
#pragma unroll
        for (int i = 0; i < 8; ++i) {
            const int srow = q0 + half * 8 + i;
            ao[((size_t)(bI * Sm + srow)) * Dm + dcol] = f2b(o[nt][i] * inv[i]);
        }
    }
}

// ---------------------------------------------------------------------------
// Host launch
// ---------------------------------------------------------------------------
extern "C" void kernel_launch(void* const* d_in, const int* in_sizes, int n_in,
                              void* d_out, int out_size, void* d_ws, size_t ws_size,
                              hipStream_t stream) {
    const float* x      = (const float*)d_in[0];
    const float* Wq     = (const float*)d_in[1];
    const float* Wk     = (const float*)d_in[2];
    const float* Wv     = (const float*)d_in[3];
    const float* Wo     = (const float*)d_in[4];
    const float* W_up   = (const float*)d_in[5];
    const float* W_down = (const float*)d_in[6];
    const float* ln1_g  = (const float*)d_in[7];
    const float* ln1_b  = (const float*)d_in[8];
    const float* ln2_g  = (const float*)d_in[9];
    const float* ln2_b  = (const float*)d_in[10];
    float* out = (float*)d_out;

    char*  ws  = (char*)d_ws;
    size_t off = 0;
    auto alloc = [&](size_t bytes) -> void* {
        void* p = ws + off;
        off = (off + bytes + 255) & ~(size_t)255;
        return p;
    };
    const size_t DD = (size_t)Dm * Dm, DF = (size_t)Dm * FFm;
    const size_t MD = (size_t)Mrows * Dm, MF = (size_t)Mrows * FFm;

    unsigned short* WqT    = (unsigned short*)alloc(DD * 2);
    unsigned short* WkT    = (unsigned short*)alloc(DD * 2);
    unsigned short* WvT    = (unsigned short*)alloc(DD * 2);
    unsigned short* WoT    = (unsigned short*)alloc(DD * 2);
    unsigned short* WupT   = (unsigned short*)alloc(DF * 2);
    unsigned short* WdownT = (unsigned short*)alloc(DF * 2);
    unsigned short* h1     = (unsigned short*)alloc(MD * 2);
    unsigned short* qb     = (unsigned short*)alloc(MD * 2);
    unsigned short* kb     = (unsigned short*)alloc(MD * 2);
    unsigned short* vtb    = (unsigned short*)alloc(MD * 2);
    unsigned short* aob    = (unsigned short*)alloc(MD * 2);
    float*          x1     = (float*)alloc(MD * 4);
    unsigned short* h2     = (unsigned short*)alloc(MD * 2);
    unsigned short* up     = (unsigned short*)alloc(MF * 2);
    (void)ws_size; (void)n_in; (void)in_sizes; (void)out_size;

    // 1) weight convert + transpose to bf16 [N][K]
    wt_bf16<<<(unsigned)(DD / 256), 256, 0, stream>>>(Wq, WqT, Dm, Dm);
    wt_bf16<<<(unsigned)(DD / 256), 256, 0, stream>>>(Wk, WkT, Dm, Dm);
    wt_bf16<<<(unsigned)(DD / 256), 256, 0, stream>>>(Wv, WvT, Dm, Dm);
    wt_bf16<<<(unsigned)(DD / 256), 256, 0, stream>>>(Wo, WoT, Dm, Dm);
    wt_bf16<<<(unsigned)(DF / 256), 256, 0, stream>>>(W_up, WupT, Dm, FFm);
    wt_bf16<<<(unsigned)(DF / 256), 256, 0, stream>>>(W_down, WdownT, FFm, Dm);

    // 2) LN1
    ln_kernel<<<Mrows, 256, 0, stream>>>(x, ln1_g, ln1_b, h1);

    // 3) QKV projections
    dim3 gDD(Dm / 128, Mrows / 128);
    gemm_bf16<1><<<gDD, 256, 0, stream>>>(h1, WqT, Dm, Dm, nullptr, qb, nullptr);
    gemm_bf16<1><<<gDD, 256, 0, stream>>>(h1, WkT, Dm, Dm, nullptr, kb, nullptr);
    gemm_bf16<2><<<gDD, 256, 0, stream>>>(h1, WvT, Dm, Dm, nullptr, vtb, nullptr);

    // 4) attention
    attn_kernel<<<dim3(Sm / 128, Bm * Hn), 256, 0, stream>>>(qb, kb, vtb, aob);

    // 5) output projection + residual -> x1 (fp32)
    gemm_bf16<3><<<gDD, 256, 0, stream>>>(aob, WoT, Dm, Dm, x, nullptr, x1);

    // 6) LN2
    ln_kernel<<<Mrows, 256, 0, stream>>>(x1, ln2_g, ln2_b, h2);

    // 7) FFN up + ReLU
    dim3 gDF(FFm / 128, Mrows / 128);
    gemm_bf16<4><<<gDF, 256, 0, stream>>>(h2, WupT, FFm, Dm, nullptr, up, nullptr);

    // 8) FFN down + residual -> out (fp32)
    gemm_bf16<3><<<gDD, 256, 0, stream>>>(up, WdownT, Dm, FFm, x1, nullptr, out);
}
